// Qwen3Attention_23605140259003
// MI455X (gfx1250) — matrix operand order repaired and checked by reading, not compiled, not run on hardware
//
#include <hip/hip_runtime.h>

// ---------------------------------------------------------------------------
// Qwen3-style attention block for MI455X (gfx1250), wave32 + WMMA bf16 +
// Tensor Data Mover (TDM) global->LDS tile DMA.
// ---------------------------------------------------------------------------

#define T_   2048
#define HID_ 4096
#define NH_  32
#define NKV_ 8
#define HD_  128
#define QKVN 6144          // (NH + 2*NKV) * HD
#define ATTN_SCALE 0.08838834764831845f   // 1/sqrt(128)

typedef __attribute__((ext_vector_type(16))) __bf16 v16bf;
typedef __attribute__((ext_vector_type(8)))  float  v8f;
typedef __attribute__((ext_vector_type(4)))  unsigned int u32x4;
typedef __attribute__((ext_vector_type(8)))  int i32x8;
typedef __attribute__((ext_vector_type(4)))  int i32x4;

union BFrag {                 // 8 VGPRs = 16 bf16 elements
  v16bf v;
  unsigned int d[8];
};

__device__ inline unsigned short f2bf(float f) {   // native RNE f32 -> bf16
  __bf16 b = (__bf16)f;
  return __builtin_bit_cast(unsigned short, b);
}

__device__ inline unsigned int pack_bf16(float lo, float hi) {
  // lets the backend emit packed f32->bf16 conversion
  return (unsigned int)f2bf(lo) | ((unsigned int)f2bf(hi) << 16);
}

__device__ inline v8f zero8() {
  v8f z;
  #pragma unroll
  for (int i = 0; i < 8; ++i) z[i] = 0.f;
  return z;
}

// A-matrix (16-bit, 16x32) K index for dword v (pairs e=2v,2v+1), lane half hl:
// VGPR0..3: K = 2v + hl*8 ; VGPR4..7: K = 16 + 2(v-4) + hl*8  (ISA 7.12.2)
__device__ inline int kmapA(int v, int hl) {
  return (v < 4 ? 2 * v : 16 + 2 * (v - 4)) + hl * 8;
}

__device__ inline v8f wmma_bf16(const BFrag& a, const BFrag& b, v8f c) {
  return __builtin_amdgcn_wmma_f32_16x16x32_bf16(
      false, a.v, false, b.v, (short)0, c, false, false);
}

// ---------------------------------------------------------------------------
// Tensor Data Mover: DMA a 2D tile of 16-bit elements (tile_d0 elems per row,
// tile_d1 rows, row stride stride0 elems) from global to LDS, with hardware
// LDS row padding (pad_interval: DWORDs between pads = 2<<code; pad_amount:
// pad DWORDs = code+1). Descriptor per CDNA5 ISA ch.8 (D# groups 0..3).
// ---------------------------------------------------------------------------
__device__ inline void tdm_load_2d(unsigned lds_byte, const void* gtile,
                                   unsigned tile_d0, unsigned tile_d1,
                                   unsigned stride0,
                                   unsigned pad_int_code, unsigned pad_amt_code) {
  unsigned long long ga = (unsigned long long)gtile;
  u32x4 g0 = { 1u,                                   // count=1 (valid D#)
               lds_byte,                              // lds_addr
               (unsigned)ga,                          // global_addr[31:0]
               (unsigned)(ga >> 32) | (2u << 30) };   // addr[56:32] | type=2
  i32x8 g1 = {
      (int)((1u << 16) |                 // data_size = 2 bytes
            (1u << 20) |                 // pad_enable
            (pad_int_code << 22) |       // pad_interval
            (pad_amt_code << 25)),       // pad_amount
      (int)((tile_d0 & 0xFFFFu) << 16),  // tensor_dim0[15:0] (tile in-bounds)
      (int)((tile_d1 & 0xFFFFu) << 16),  // tensor_dim0 hi=0 | tensor_dim1 lo
      (int)((tile_d0 & 0xFFFFu) << 16),  // tensor_dim1 hi=0 | tile_dim0
      (int)(tile_d1 & 0xFFFFu),          // tile_dim1 | tile_dim2=0
      (int)stride0,                      // tensor_dim0_stride[31:0]
      0, 0 };                            // stride hi, dim1_stride = 0
  i32x4 z4 = {0, 0, 0, 0};
#if __has_include(<hip/amd_detail/amd_gfx1250_TDM.h>)
  i32x8 z8 = {0, 0, 0, 0, 0, 0, 0, 0};
  __builtin_amdgcn_tensor_load_to_lds(g0, g1, z4, z4, z8, 0);
#else
  __builtin_amdgcn_tensor_load_to_lds(g0, g1, z4, z4, 0);
#endif
}

// ---------------------------------------------------------------------------
// Kernel 1: x = lnL^T @ reshape(hidden) @ lnR, then per-token sym int8 fake
// quant, stored as bf16 activations.
// ---------------------------------------------------------------------------
__global__ __launch_bounds__(256) void k_transform_quant(
    const float* __restrict__ hs, const float* __restrict__ lnL,
    const float* __restrict__ lnR, unsigned short* __restrict__ xb) {
  const int t = blockIdx.x, tid = threadIdx.x;
  __shared__ float xs[4096];
  __shared__ float ts[4096];
  __shared__ float red[256];
  for (int i = tid; i < 4096; i += 256) xs[i] = hs[(size_t)t * HID_ + i];
  __syncthreads();
  for (int i = tid; i < 4096; i += 256) {
    int L = i >> 6, r = i & 63;
    float a = 0.f;
    #pragma unroll 8
    for (int l = 0; l < 64; ++l) a = fmaf(lnL[l * 64 + L], xs[l * 64 + r], a);
    ts[i] = a;
  }
  __syncthreads();
  float y[16];
  float am = 0.f;
  #pragma unroll
  for (int c = 0; c < 16; ++c) {
    int i = tid + c * 256;
    int L = i >> 6, R = i & 63;
    float a = 0.f;
    #pragma unroll 8
    for (int r = 0; r < 64; ++r) a = fmaf(ts[L * 64 + r], lnR[r * 64 + R], a);
    y[c] = a;
    am = fmaxf(am, fabsf(a));
  }
  red[tid] = am;
  __syncthreads();
  for (int s = 128; s > 0; s >>= 1) {
    if (tid < s) red[tid] = fmaxf(red[tid], red[tid + s]);
    __syncthreads();
  }
  float scale = fmaxf(red[0] * (1.f / 127.f), 1e-8f);
  #pragma unroll
  for (int c = 0; c < 16; ++c) {
    int i = tid + c * 256;
    float q = fminf(fmaxf(rintf(y[c] / scale), -128.f), 127.f);
    xb[(size_t)t * HID_ + i] = f2bf(q * scale);
  }
}

// ---------------------------------------------------------------------------
// Kernel 2/8: C(MxN,f32) = A(MxK,bf16) @ B(NxK,f32)^T.
// A tile (128x32 bf16, padded rows) DMA'd by the TDM; B tile converted
// f32->bf16 (native packed cvt) while staging to LDS. 8 waves x (4x2) WMMA.
// ---------------------------------------------------------------------------
__global__ __launch_bounds__(256) void k_gemm_bf16w(
    const unsigned short* __restrict__ A, const float* __restrict__ B,
    float* __restrict__ C, int M, int N, int K) {
  const int nblk = N >> 7;
  const int bx = blockIdx.x % nblk;
  const int by = blockIdx.x / nblk;
  const int tid = threadIdx.x;
  const int w = tid >> 5, lane = tid & 31;
  const int wm = w >> 2, wn = w & 3;
  const int hl = lane >> 4, l16 = lane & 15;

  __shared__ unsigned short As[128][40];   // 32 data + 8 pad (16 DW + 4 DW)
  __shared__ unsigned short Bs[128][40];

  v8f acc[4][2];
  #pragma unroll
  for (int i = 0; i < 4; ++i)
    #pragma unroll
    for (int j = 0; j < 2; ++j) acc[i][j] = zero8();

  const unsigned as_lds = (unsigned)(size_t)&As[0][0];  // flat low 32 = LDS offset
  const unsigned short* Atile = A + (size_t)(by * 128) * K;
  const int lrow = tid >> 1, lseg = (tid & 1) * 16;
  const float* Bg = B + (size_t)(bx * 128 + lrow) * K + lseg;

  for (int k0 = 0; k0 < K; k0 += 32) {
    // TDM: A tile 128 rows x 32 bf16, row stride K, LDS rows padded 16DW+4DW
    if (w == 0)
      tdm_load_2d(as_lds, Atile + k0, 32u, 128u, (unsigned)K, 3u, 3u);
    // stage B (f32 -> bf16): 16 floats / thread
    #pragma unroll
    for (int i = 0; i < 4; ++i) {
      float4 bv = *(const float4*)(Bg + k0 + 4 * i);
      unsigned int* bsd = (unsigned int*)&Bs[lrow][lseg + 4 * i];
      bsd[0] = pack_bf16(bv.x, bv.y);
      bsd[1] = pack_bf16(bv.z, bv.w);
    }
    if (w == 0) __builtin_amdgcn_s_wait_tensorcnt(0);
    __syncthreads();

    BFrag bf[2];
    #pragma unroll
    for (int ns = 0; ns < 2; ++ns) {
      int brow = wn * 32 + ns * 16 + l16;
      #pragma unroll
      for (int v = 0; v < 8; ++v)  // B layout: lane n=l16, K = kmapA(v, hl) (+0/1)
        bf[ns].d[v] = *(const unsigned int*)&Bs[brow][kmapA(v, hl)];
    }
    #pragma unroll
    for (int ms = 0; ms < 4; ++ms) {
      BFrag af;
      int arow = wm * 64 + ms * 16 + l16;
      #pragma unroll
      for (int v = 0; v < 8; ++v)
        af.d[v] = *(const unsigned int*)&As[arow][kmapA(v, hl)];
      #pragma unroll
      for (int ns = 0; ns < 2; ++ns)
        acc[ms][ns] = wmma_bf16(af, bf[ns], acc[ms][ns]);
    }
    __syncthreads();
  }

  // C layout: lane n = l16, row = r + 8*hl
  #pragma unroll
  for (int ms = 0; ms < 4; ++ms)
    #pragma unroll
    for (int ns = 0; ns < 2; ++ns) {
      int row = by * 128 + wm * 64 + ms * 16 + hl * 8;
      int col = bx * 128 + wn * 32 + ns * 16 + l16;
      #pragma unroll
      for (int r = 0; r < 8; ++r)
        C[(size_t)(row + r) * N + col] = acc[ms][ns][r];
    }
}

// ---------------------------------------------------------------------------
// Kernels 3-5: per-vector post-processing of qkv.
// mode 0 (q): rmsnorm + rope + @k_trans          (inv(k_trans).T == k_trans)
// mode 1 (k): rmsnorm + rope + @k_trans + asym4 fake quant (group = HD)
// mode 2 (v): asym4 fake quant only
// ---------------------------------------------------------------------------
__global__ __launch_bounds__(128) void k_qkv_post(
    const float* __restrict__ qkv, const float* __restrict__ normw,
    const float* __restrict__ ktrans, const int* __restrict__ positions,
    unsigned short* __restrict__ out, int nh, int off, int mode) {
  const int t = blockIdx.x / nh;
  const int h = blockIdx.x % nh;
  const int d = threadIdx.x;
  __shared__ float sv[128];
  __shared__ float r0[128];
  __shared__ float r1[128];

  float x = qkv[(size_t)t * QKVN + off + h * HD_ + d];

  if (mode <= 1) {
    r0[d] = x * x;
    __syncthreads();
    for (int s = 64; s > 0; s >>= 1) {
      if (d < s) r0[d] += r0[d + s];
      __syncthreads();
    }
    float y = x * rsqrtf(r0[0] * (1.f / 128.f) + 1e-6f) * normw[d];
    __syncthreads();
    sv[d] = y;
    __syncthreads();
    // rope
    float pos = (float)positions[t];
    int j = d & 63;
    float fr = pos * __expf(-((float)j * (1.f / 64.f)) * 13.815510557964274f);
    float cs = __cosf(fr), sn = __sinf(fr);
    float x1 = sv[j], x2 = sv[j + 64];
    float rr = (d < 64) ? (x1 * cs - x2 * sn) : (x2 * cs + x1 * sn);
    __syncthreads();
    sv[d] = rr;
    __syncthreads();
    // @ k_trans (128x128)
    float acc = 0.f;
    #pragma unroll 8
    for (int dd = 0; dd < 128; ++dd) acc = fmaf(sv[dd], ktrans[dd * 128 + d], acc);
    x = acc;
    __syncthreads();
  }
  if (mode >= 1) {
    r0[d] = x;
    r1[d] = x;
    __syncthreads();
    for (int s = 64; s > 0; s >>= 1) {
      if (d < s) {
        r0[d] = fminf(r0[d], r0[d + s]);
        r1[d] = fmaxf(r1[d], r1[d + s]);
      }
      __syncthreads();
    }
    float xmin = fminf(r0[0], 0.f), xmax = fmaxf(r1[0], 0.f);
    float scale = fmaxf((xmax - xmin) * (1.f / 15.f), 1e-8f);
    float zero = rintf(-xmin / scale);
    float q = fminf(fmaxf(rintf(x / scale) + zero, 0.f), 15.f);
    x = (q - zero) * scale;
  }
  out[((size_t)t * nh + h) * HD_ + d] = f2bf(x);
}

// ---------------------------------------------------------------------------
// Kernel 6: causal flash attention, GQA 32q/8kv, HD=128, bf16 WMMA.
// One block = (head, 128-query tile); each wave owns 16 query rows.
// K/V 32x128 tiles are TDM-DMA'd to LDS (padded rows: 64 DW + 4 DW) once per
// block instead of 8 waves gathering the same data. Per 32-key block:
// 8 WMMA (S = Q K^T), online softmax, 8 WMMA (O += P V).
// ---------------------------------------------------------------------------
__global__ __launch_bounds__(256) void k_attn(
    const unsigned short* __restrict__ Q, const unsigned short* __restrict__ Kc,
    const unsigned short* __restrict__ Vc, float* __restrict__ O) {
  const int mblocks = T_ / 128;
  const int h = blockIdx.x / mblocks;
  const int mb = blockIdx.x % mblocks;
  const int kvh = h >> 2;  // NH/NKV = 4
  const int tid = threadIdx.x, w = tid >> 5, lane = tid & 31;
  const int hl = lane >> 4, l16 = lane & 15;
  const int qr0 = mb * 128 + w * 16;

  __shared__ unsigned short Pl[8][16][32];   // per-wave P staging (8 KB)
  __shared__ unsigned short Kl[32][136];     // 128 data + 8 pad ushorts
  __shared__ unsigned short Vl[32][136];

  const unsigned kl_lds = (unsigned)(size_t)&Kl[0][0];
  const unsigned vl_lds = (unsigned)(size_t)&Vl[0][0];

  // Q fragments: 16 rows x 128 head-dims = 4 A-frags (held in VGPRs)
  BFrag qf[4];
  #pragma unroll
  for (int kt = 0; kt < 4; ++kt)
    #pragma unroll
    for (int v = 0; v < 8; ++v) {
      int kd = kt * 32 + kmapA(v, hl);
      qf[kt].d[v] =
          *(const unsigned int*)&Q[((size_t)(qr0 + l16) * NH_ + h) * HD_ + kd];
    }

  float mrow[8], lrowv[8];
  v8f o[8];
  #pragma unroll
  for (int r = 0; r < 8; ++r) { mrow[r] = -1e30f; lrowv[r] = 0.f; }
  #pragma unroll
  for (int dt = 0; dt < 8; ++dt) o[dt] = zero8();

  const int njb = mb * 4 + 4;  // 32-key blocks up to the causal frontier
  for (int j = 0; j < njb; ++j) {
    if (w == 0) {
      const unsigned short* kt0 = Kc + ((size_t)(j * 32) * NKV_ + kvh) * HD_;
      const unsigned short* vt0 = Vc + ((size_t)(j * 32) * NKV_ + kvh) * HD_;
      tdm_load_2d(kl_lds, kt0, 128u, 32u, (unsigned)(NKV_ * HD_), 5u, 3u);
      tdm_load_2d(vl_lds, vt0, 128u, 32u, (unsigned)(NKV_ * HD_), 5u, 3u);
      __builtin_amdgcn_s_wait_tensorcnt(0);
    }
    __syncthreads();

    v8f s0 = zero8(), s1 = zero8();
    #pragma unroll
    for (int kt = 0; kt < 4; ++kt) {
      BFrag b0, b1;
      #pragma unroll
      for (int v = 0; v < 8; ++v) {  // B layout: lane n = key, K = kmapA(v, hl) (+0/1)
        int kd = kt * 32 + kmapA(v, hl);
        b0.d[v] = *(const unsigned int*)&Kl[l16][kd];
        b1.d[v] = *(const unsigned int*)&Kl[l16 + 16][kd];
      }
      s0 = wmma_bf16(qf[kt], b0, s0);
      s1 = wmma_bf16(qf[kt], b1, s1);
    }

    float p0[8], p1[8], mx[8];
    #pragma unroll
    for (int r = 0; r < 8; ++r) {
      int row = qr0 + r + 8 * hl;
      int c0 = j * 32 + l16, c1 = c0 + 16;
      p0[r] = (c0 <= row) ? s0[r] * ATTN_SCALE : -1e30f;
      p1[r] = (c1 <= row) ? s1[r] * ATTN_SCALE : -1e30f;
      mx[r] = fmaxf(p0[r], p1[r]);
    }
    // row-max across the 16 lanes of each half (xor<16 keeps halves separate)
    #pragma unroll
    for (int off = 8; off >= 1; off >>= 1)
      #pragma unroll
      for (int r = 0; r < 8; ++r)
        mx[r] = fmaxf(mx[r], __shfl_xor(mx[r], off, 32));

    float corr[8], sm[8];
    #pragma unroll
    for (int r = 0; r < 8; ++r) {
      float mn = fmaxf(mrow[r], mx[r]);
      corr[r] = __expf(mrow[r] - mn);
      mrow[r] = mn;
      p0[r] = __expf(p0[r] - mn);
      p1[r] = __expf(p1[r] - mn);
      sm[r] = p0[r] + p1[r];
    }
    #pragma unroll
    for (int off = 8; off >= 1; off >>= 1)
      #pragma unroll
      for (int r = 0; r < 8; ++r) sm[r] += __shfl_xor(sm[r], off, 32);
    #pragma unroll
    for (int r = 0; r < 8; ++r) lrowv[r] = lrowv[r] * corr[r] + sm[r];
    #pragma unroll
    for (int dt = 0; dt < 8; ++dt)
      #pragma unroll
      for (int r = 0; r < 8; ++r) o[dt][r] *= corr[r];

    // stage P through per-wave LDS: C layout -> A layout
    #pragma unroll
    for (int r = 0; r < 8; ++r) {
      Pl[w][r + 8 * hl][l16] = f2bf(p0[r]);
      Pl[w][r + 8 * hl][l16 + 16] = f2bf(p1[r]);
    }
    BFrag pf;
    #pragma unroll
    for (int v = 0; v < 8; ++v)
      pf.d[v] = *(const unsigned int*)&Pl[w][l16][kmapA(v, hl)];

    // O += P @ V  (V frags from padded LDS tile)
    #pragma unroll
    for (int dt = 0; dt < 8; ++dt) {
      BFrag vf;
      #pragma unroll
      for (int v = 0; v < 8; ++v) {
        int key = kmapA(v, hl);
        unsigned int u0 = Vl[key][dt * 16 + l16];
        unsigned int u1 = Vl[key + 1][dt * 16 + l16];
        vf.d[v] = u0 | (u1 << 16);
      }
      o[dt] = wmma_bf16(pf, vf, o[dt]);
    }
    __syncthreads();   // K/V tiles consumed; safe for next TDM to overwrite
  }

  #pragma unroll
  for (int dt = 0; dt < 8; ++dt)
    #pragma unroll
    for (int r = 0; r < 8; ++r)
      O[((size_t)(qr0 + r + 8 * hl) * NH_ + h) * HD_ + dt * 16 + l16] =
          o[dt][r] / lrowv[r];
}

// ---------------------------------------------------------------------------
// Kernel 7: out[t,i,d] = sum_j o_trans[j,i] * attn[t,j,d]; sym int8 fake
// quant per token row of 4096; store bf16.
// ---------------------------------------------------------------------------
__global__ __launch_bounds__(256) void k_otrans_quant(
    const float* __restrict__ attn, const float* __restrict__ otr,
    unsigned short* __restrict__ ob) {
  const int t = blockIdx.x, tid = threadIdx.x;
  __shared__ float as[4096];
  __shared__ float red[256];
  for (int i = tid; i < 4096; i += 256) as[i] = attn[(size_t)t * 4096 + i];
  __syncthreads();
  float y[16];
  float am = 0.f;
  #pragma unroll
  for (int c = 0; c < 16; ++c) {
    int i = tid + c * 256;
    int hi = i >> 7, d = i & 127;
    float a = 0.f;
    #pragma unroll 8
    for (int jh = 0; jh < 32; ++jh)
      a = fmaf(otr[jh * 32 + hi], as[jh * 128 + d], a);
    y[c] = a;
    am = fmaxf(am, fabsf(a));
  }
  red[tid] = am;
  __syncthreads();
  for (int s = 128; s > 0; s >>= 1) {
    if (tid < s) red[tid] = fmaxf(red[tid], red[tid + s]);
    __syncthreads();
  }
  float scale = fmaxf(red[0] * (1.f / 127.f), 1e-8f);
  #pragma unroll
  for (int c = 0; c < 16; ++c) {
    int i = tid + c * 256;
    float q = fminf(fmaxf(rintf(y[c] / scale), -128.f), 127.f);
    ob[(size_t)t * 4096 + i] = f2bf(q * scale);
  }
}

// ---------------------------------------------------------------------------
extern "C" void kernel_launch(void* const* d_in, const int* in_sizes, int n_in,
                              void* d_out, int out_size, void* d_ws,
                              size_t ws_size, hipStream_t stream) {
  const float* hidden = (const float*)d_in[0];
  const int* positions = (const int*)d_in[1];
  const float* w_qkv = (const float*)d_in[2];
  const float* w_o = (const float*)d_in[3];
  const float* q_norm_w = (const float*)d_in[4];
  const float* k_norm_w = (const float*)d_in[5];
  const float* ln_left = (const float*)d_in[6];
  const float* ln_right = (const float*)d_in[7];
  const float* o_trans = (const float*)d_in[8];
  const float* k_trans = (const float*)d_in[9];
  float* out = (float*)d_out;

  char* ws = (char*)d_ws;
  unsigned short* xb = (unsigned short*)(ws + 0);           // 16 MB  (bf16 x)
  float* qkv = (float*)(ws + 16777216);                     // 48 MB  (f32 qkv)
  unsigned short* qb = (unsigned short*)(ws + 67108864);    // 16 MB
  unsigned short* kb = (unsigned short*)(ws + 83886080);    //  4 MB
  unsigned short* vb = (unsigned short*)(ws + 88080384);    //  4 MB
  float* attn = (float*)(ws + 92274688);                    // 32 MB
  unsigned short* ob = xb;  // xb is dead after the QKV GEMM; reuse region

  k_transform_quant<<<T_, 256, 0, stream>>>(hidden, ln_left, ln_right, xb);
  k_gemm_bf16w<<<(T_ / 128) * (QKVN / 128), 256, 0, stream>>>(
      xb, w_qkv, qkv, T_, QKVN, HID_);
  k_qkv_post<<<T_ * NH_, 128, 0, stream>>>(qkv, q_norm_w, k_trans, positions,
                                           qb, NH_, 0, 0);
  k_qkv_post<<<T_ * NKV_, 128, 0, stream>>>(qkv, k_norm_w, k_trans, positions,
                                            kb, NKV_, NH_ * HD_, 1);
  k_qkv_post<<<T_ * NKV_, 128, 0, stream>>>(qkv, k_norm_w, k_trans, positions,
                                            vb, NKV_, (NH_ + NKV_) * HD_, 2);
  k_attn<<<NH_ * (T_ / 128), 256, 0, stream>>>(qb, kb, vb, attn);
  k_otrans_quant<<<T_, 256, 0, stream>>>(attn, o_trans, ob);
  k_gemm_bf16w<<<(T_ / 128) * (HID_ / 128), 256, 0, stream>>>(
      ob, w_o, out, T_, HID_, HID_);
}
